// SparseMoE_20091857010896
// MI455X (gfx1250) — compile-verified, matrix operand order checked by reading, not run on hardware
//
#include <hip/hip_runtime.h>

#define B_    4
#define T_    2048
#define BT_   8192
#define D_    1024
#define H_    4096
#define E_    8
#define CAP_  8192   // per-expert token capacity (each token contributes at most once)

typedef __attribute__((ext_vector_type(16))) __bf16 bf16x16;
typedef __attribute__((ext_vector_type(8)))  float  f32x8;

__device__ __forceinline__ __bf16 f2bf(float f) {
  unsigned int u = __builtin_bit_cast(unsigned int, f);
  u += 0x7fffu + ((u >> 16) & 1u);              // round-to-nearest-even
  unsigned short h = (unsigned short)(u >> 16);
  return __builtin_bit_cast(__bf16, h);
}

// 16-bit A/B fragment layout (16x32, wave32): lane L holds row (L&15);
// per lane the 16 bf16 elements are K = 8*(L>>4)+j (j=0..7) then K = 16+8*(L>>4)+j.
__device__ __forceinline__ bf16x16 frag_from_lds(const __bf16* row, int g) {
  bf16x16 v;
  const __bf16* p0 = row + 8 * g;
  const __bf16* p1 = row + 16 + 8 * g;
#pragma unroll
  for (int j = 0; j < 8; ++j) { v[j] = p0[j]; v[j + 8] = p1[j]; }
  return v;
}

__global__ __launch_bounds__(256) void zero_kernel(float* __restrict__ out, int n,
                                                   int* __restrict__ counts) {
  int i = blockIdx.x * blockDim.x + threadIdx.x;
  if (i < n) out[i] = 0.f;
  if (i < E_) counts[i] = 0;
}

__global__ __launch_bounds__(256) void router_kernel(
    const float* __restrict__ x, const float* __restrict__ noise,
    const float* __restrict__ Wr, const float* __restrict__ Wn,
    int* __restrict__ counts, int* __restrict__ tok_list,
    float* __restrict__ gate_list) {
  const int lane = threadIdx.x & 31;
  const int wv   = threadIdx.x >> 5;
  const int tok  = blockIdx.x * 8 + wv;
  if (tok >= BT_) return;

  float ar[E_], an[E_];
#pragma unroll
  for (int e = 0; e < E_; ++e) { ar[e] = 0.f; an[e] = 0.f; }

  const float* xr = x + (size_t)tok * D_;
  for (int i = 0; i < D_ / 32; ++i) {
    const int d = i * 32 + lane;                 // coalesced across lanes
    const float xv = xr[d];
#pragma unroll
    for (int e = 0; e < E_; ++e) {
      ar[e] = fmaf(xv, Wr[e * D_ + d], ar[e]);
      an[e] = fmaf(xv, Wn[e * D_ + d], an[e]);
    }
  }
#pragma unroll
  for (int e = 0; e < E_; ++e) {
#pragma unroll
    for (int off = 16; off > 0; off >>= 1) {
      ar[e] += __shfl_xor(ar[e], off, 32);
      an[e] += __shfl_xor(an[e], off, 32);
    }
  }
  if (lane == 0) {
    float noisy[E_];
#pragma unroll
    for (int e = 0; e < E_; ++e) {
      const float nl = an[e];
      const float sp = (nl > 20.f) ? nl : log1pf(__expf(nl));   // softplus
      noisy[e] = ar[e] + noise[tok * E_ + e] * sp;
    }
    int i0 = 0;
#pragma unroll
    for (int e = 1; e < E_; ++e) if (noisy[e] > noisy[i0]) i0 = e;
    int i1 = (i0 == 0) ? 1 : 0;
#pragma unroll
    for (int e = 0; e < E_; ++e)
      if (e != i0 && noisy[e] > noisy[i1]) i1 = e;
    const float mx = fmaxf(noisy[i0], noisy[i1]);
    const float e0 = __expf(noisy[i0] - mx);
    const float e1 = __expf(noisy[i1] - mx);
    const float inv = 1.f / (e0 + e1);
    const int s0 = atomicAdd(&counts[i0], 1);
    tok_list[i0 * CAP_ + s0]  = tok;
    gate_list[i0 * CAP_ + s0] = e0 * inv;
    const int s1 = atomicAdd(&counts[i1], 1);
    tok_list[i1 * CAP_ + s1]  = tok;
    gate_list[i1 * CAP_ + s1] = e1 * inv;
  }
}

__global__ void scan_kernel(const int* __restrict__ counts, int* __restrict__ offsets) {
  if (threadIdx.x == 0 && blockIdx.x == 0) {
    int acc = 0;
    for (int e = 0; e < E_; ++e) { offsets[e] = acc; acc += counts[e]; }
  }
}

// ---------------- fc1: h = relu(x_gathered @ W1[e])^2 (bf16 out) ----------------
__global__ __launch_bounds__(256) void fc1_kernel(
    const float* __restrict__ x, const float* __restrict__ W1,
    const int* __restrict__ counts, const int* __restrict__ offsets,
    const int* __restrict__ tok_list, __bf16* __restrict__ hbuf) {
  const int e   = blockIdx.z;
  const int cnt = counts[e];
  const int m0  = blockIdx.y * 128;
  if (m0 >= cnt) return;
  const int n0  = blockIdx.x * 128;
  const float* __restrict__ W = W1 + (size_t)e * D_ * H_;

  __shared__ __align__(16) __bf16 As[128][40];   // [row][k], padded
  __shared__ __align__(16) __bf16 Bt[128][40];   // transposed W tile: [n][k]
  __shared__ int toks[128];

  const int t = threadIdx.x;
  if (t < 128) {
    int r = m0 + t;
    if (r >= cnt) r = cnt - 1;                   // clamp (duplicates harmless)
    toks[t] = tok_list[e * CAP_ + r];
  }
  __syncthreads();

  const int lane  = t & 31;
  const int w     = t >> 5;
  const int waveM = (w & 3) * 32;                // 4 waves along M
  const int waveN = (w >> 2) * 64;               // 2 waves along N
  const int g     = lane >> 4;
  const int nr    = lane & 15;

  f32x8 acc[2][4];
#pragma unroll
  for (int mf = 0; mf < 2; ++mf)
#pragma unroll
    for (int nf = 0; nf < 4; ++nf)
#pragma unroll
      for (int i = 0; i < 8; ++i) acc[mf][nf][i] = 0.f;

  const int arow = t & 127;
  const int acb  = (t >> 7) * 16;                // 0 or 16
  const int brow = t >> 3;                       // 0..31
  const int bcb  = (t & 7) * 16;                 // 0..112
  const size_t xrow = (size_t)toks[arow] * D_;

  for (int k0 = 0; k0 < D_; k0 += 32) {
    {   // A tile: gathered x rows, fp32 -> bf16
      const float4* xp = (const float4*)(x + xrow + k0 + acb);
#pragma unroll
      for (int q = 0; q < 4; ++q) {
        float4 v = xp[q];
        As[arow][acb + q * 4 + 0] = f2bf(v.x);
        As[arow][acb + q * 4 + 1] = f2bf(v.y);
        As[arow][acb + q * 4 + 2] = f2bf(v.z);
        As[arow][acb + q * 4 + 3] = f2bf(v.w);
      }
    }
    {   // B tile: W1 rows k0..k0+31, cols n0..n0+127; store transposed
      const float4* wp = (const float4*)(W + (size_t)(k0 + brow) * H_ + n0 + bcb);
#pragma unroll
      for (int q = 0; q < 4; ++q) {
        float4 v = wp[q];
        Bt[bcb + q * 4 + 0][brow] = f2bf(v.x);
        Bt[bcb + q * 4 + 1][brow] = f2bf(v.y);
        Bt[bcb + q * 4 + 2][brow] = f2bf(v.z);
        Bt[bcb + q * 4 + 3][brow] = f2bf(v.w);
      }
    }
    if (k0 + 32 < D_) {
      __builtin_prefetch(x + xrow + k0 + 32 + acb, 0, 0);
      __builtin_prefetch(W + (size_t)(k0 + 32 + brow) * H_ + n0 + bcb, 0, 0);
    }
    __syncthreads();

    bf16x16 afrag[2], bfrag[4];
#pragma unroll
    for (int mf = 0; mf < 2; ++mf)
      afrag[mf] = frag_from_lds(&As[waveM + mf * 16 + nr][0], g);
#pragma unroll
    for (int nf = 0; nf < 4; ++nf)
      bfrag[nf] = frag_from_lds(&Bt[waveN + nf * 16 + nr][0], g);
#pragma unroll
    for (int mf = 0; mf < 2; ++mf)
#pragma unroll
      for (int nf = 0; nf < 4; ++nf)
        acc[mf][nf] = __builtin_amdgcn_wmma_f32_16x16x32_bf16(
            false, afrag[mf], false, bfrag[nf], (short)0, acc[mf][nf], false, false);
    __syncthreads();
  }

  const int hbase = offsets[e] + m0;
#pragma unroll
  for (int mf = 0; mf < 2; ++mf) {
#pragma unroll
    for (int nf = 0; nf < 4; ++nf) {
      const int col = n0 + waveN + nf * 16 + nr;
#pragma unroll
      for (int i = 0; i < 8; ++i) {
        const int ml = waveM + mf * 16 + i + 8 * g;   // C/D layout: VGPR i -> M=i / i+8
        if (m0 + ml < cnt) {
          float v = fmaxf(acc[mf][nf][i], 0.f);
          hbuf[(size_t)(hbase + ml) * H_ + col] = f2bf(v * v);   // relu^2
        }
      }
    }
  }
}

// ---------------- fc2: out[tok] += gate * (h @ W2[e]) ----------------
__global__ __launch_bounds__(256) void fc2_kernel(
    const __bf16* __restrict__ hbuf, const float* __restrict__ W2,
    const int* __restrict__ counts, const int* __restrict__ offsets,
    const int* __restrict__ tok_list, const float* __restrict__ gate_list,
    float* __restrict__ out) {
  const int e   = blockIdx.z;
  const int cnt = counts[e];
  const int m0  = blockIdx.y * 128;
  if (m0 >= cnt) return;
  const int n0  = blockIdx.x * 128;
  const float* __restrict__ W = W2 + (size_t)e * H_ * D_;

  __shared__ __align__(16) __bf16 As[128][40];
  __shared__ __align__(16) __bf16 Bt[128][40];
  __shared__ int   toks[128];
  __shared__ float gts[128];

  const int t = threadIdx.x;
  if (t < 128) {
    int r = m0 + t;
    if (r >= cnt) r = cnt - 1;
    toks[t] = tok_list[e * CAP_ + r];
    gts[t]  = gate_list[e * CAP_ + r];
  }
  __syncthreads();

  const int lane  = t & 31;
  const int w     = t >> 5;
  const int waveM = (w & 3) * 32;
  const int waveN = (w >> 2) * 64;
  const int g     = lane >> 4;
  const int nr    = lane & 15;

  f32x8 acc[2][4];
#pragma unroll
  for (int mf = 0; mf < 2; ++mf)
#pragma unroll
    for (int nf = 0; nf < 4; ++nf)
#pragma unroll
      for (int i = 0; i < 8; ++i) acc[mf][nf][i] = 0.f;

  const int arow = t & 127;
  const int acb  = (t >> 7) * 16;
  int ar = m0 + arow; if (ar >= cnt) ar = cnt - 1;
  const size_t hrow = (size_t)(offsets[e] + ar) * H_;
  const int brow = t >> 3;
  const int bcb  = (t & 7) * 16;

  for (int k0 = 0; k0 < H_; k0 += 32) {
    {   // A tile: h already bf16, copy 32B per thread
      const uint4* hp = (const uint4*)(hbuf + hrow + k0 + acb);
      uint4* dp = (uint4*)&As[arow][acb];
      dp[0] = hp[0];
      dp[1] = hp[1];
    }
    {   // B tile: W2 rows k0..k0+31 (H dim), cols n0..n0+127 (D dim), transposed
      const float4* wp = (const float4*)(W + (size_t)(k0 + brow) * D_ + n0 + bcb);
#pragma unroll
      for (int q = 0; q < 4; ++q) {
        float4 v = wp[q];
        Bt[bcb + q * 4 + 0][brow] = f2bf(v.x);
        Bt[bcb + q * 4 + 1][brow] = f2bf(v.y);
        Bt[bcb + q * 4 + 2][brow] = f2bf(v.z);
        Bt[bcb + q * 4 + 3][brow] = f2bf(v.w);
      }
    }
    if (k0 + 32 < H_) {
      __builtin_prefetch(hbuf + hrow + k0 + 32 + acb, 0, 0);
      __builtin_prefetch(W + (size_t)(k0 + 32 + brow) * D_ + n0 + bcb, 0, 0);
    }
    __syncthreads();

    bf16x16 afrag[2], bfrag[4];
#pragma unroll
    for (int mf = 0; mf < 2; ++mf)
      afrag[mf] = frag_from_lds(&As[waveM + mf * 16 + nr][0], g);
#pragma unroll
    for (int nf = 0; nf < 4; ++nf)
      bfrag[nf] = frag_from_lds(&Bt[waveN + nf * 16 + nr][0], g);
#pragma unroll
    for (int mf = 0; mf < 2; ++mf)
#pragma unroll
      for (int nf = 0; nf < 4; ++nf)
        acc[mf][nf] = __builtin_amdgcn_wmma_f32_16x16x32_bf16(
            false, afrag[mf], false, bfrag[nf], (short)0, acc[mf][nf], false, false);
    __syncthreads();
  }

#pragma unroll
  for (int mf = 0; mf < 2; ++mf) {
#pragma unroll
    for (int nf = 0; nf < 4; ++nf) {
      const int col = n0 + waveN + nf * 16 + nr;
#pragma unroll
      for (int i = 0; i < 8; ++i) {
        const int ml = waveM + mf * 16 + i + 8 * g;
        if (m0 + ml < cnt) {
          const float v = acc[mf][nf][i] * gts[ml];
          __hip_atomic_fetch_add(&out[(size_t)toks[ml] * D_ + col], v,
                                 __ATOMIC_RELAXED, __HIP_MEMORY_SCOPE_AGENT);
        }
      }
    }
  }
}

extern "C" void kernel_launch(void* const* d_in, const int* in_sizes, int n_in,
                              void* d_out, int out_size, void* d_ws, size_t ws_size,
                              hipStream_t stream) {
  const float* x     = (const float*)d_in[0];
  const float* noise = (const float*)d_in[1];
  const float* Wr    = (const float*)d_in[2];
  const float* Wn    = (const float*)d_in[3];
  const float* W1    = (const float*)d_in[4];
  const float* W2    = (const float*)d_in[5];
  float* out = (float*)d_out;

  // workspace layout
  char* ws = (char*)d_ws;
  int*    counts    = (int*)ws;                                   // 8
  int*    offsets   = counts + 8;                                 // 8
  int*    tok_list  = (int*)(ws + 1024);                          // E*CAP ints (256 KB)
  float*  gate_list = (float*)(ws + 1024 + E_ * CAP_ * 4);        // E*CAP floats (256 KB)
  __bf16* hbuf      = (__bf16*)(ws + (1 << 20));                  // 16384 x 4096 bf16 (128 MB)

  {
    const int n = out_size;                                       // B*T*D
    zero_kernel<<<(n + 255) / 256, 256, 0, stream>>>(out, n, counts);
  }
  router_kernel<<<BT_ / 8, 256, 0, stream>>>(x, noise, Wr, Wn, counts, tok_list, gate_list);
  scan_kernel<<<1, 32, 0, stream>>>(counts, offsets);
  fc1_kernel<<<dim3(H_ / 128, BT_ / 128, E_), 256, 0, stream>>>(
      x, W1, counts, offsets, tok_list, hbuf);
  fc2_kernel<<<dim3(D_ / 128, BT_ / 128, E_), 256, 0, stream>>>(
      hbuf, W2, counts, offsets, tok_list, gate_list, out);
}
